// RowAttentionWithPairBias_7215545057291
// MI455X (gfx1250) — compile-verified, matrix operand order checked by reading, not run on hardware
//
#include <hip/hip_runtime.h>
#include <hip/hip_bf16.h>
#include <math.h>

typedef __bf16 bf16;
typedef __attribute__((ext_vector_type(16))) __bf16 v16bf;
typedef __attribute__((ext_vector_type(8)))  __bf16 v8bf;
typedef __attribute__((ext_vector_type(8)))  float  v8f;

static __device__ __forceinline__ v8f wmma_bf16(v16bf a, v16bf b, v8f c) {
  return __builtin_amdgcn_wmma_f32_16x16x32_bf16(false, a, false, b, (short)0, c, false, false);
}

// Async copy engine: memory -> LDS, 16B per lane, tracked by ASYNCcnt.
static __device__ __forceinline__ void async_ld_b128(unsigned lds_addr, const void* g) {
  asm volatile("global_load_async_to_lds_b128 %0, %1, off"
               :: "v"(lds_addr), "v"((unsigned long long)(uintptr_t)g)
               : "memory");
}
static __device__ __forceinline__ void wait_async0() {
  asm volatile("s_wait_asynccnt 0x0" ::: "memory");
}
// Low 32 bits of a generic LDS pointer == byte offset in the workgroup's LDS.
static __device__ __forceinline__ unsigned lds_off(const void* p) {
  return (unsigned)(uintptr_t)p;
}

// 16x32 bf16 fragment (A layout; also B when B-columns are stored as rows).
// lanes 0-15: row = lane, K = {0..7,16..23}; lanes 16-31: row = lane-16, K = {8..15,24..31}.
static __device__ __forceinline__ v16bf load_frag(const bf16* base, int ld, int k0) {
  int lane = threadIdx.x & 31;
  const bf16* p = base + (size_t)(lane & 15) * ld + k0 + ((lane >> 4) << 3);
  v8bf a0 = *(const v8bf*)(p);
  v8bf a1 = *(const v8bf*)(p + 16);
  v16bf f;
#pragma unroll
  for (int i = 0; i < 8; ++i) { f[i] = a0[i]; f[i + 8] = a1[i]; }
  return f;
}

static __device__ __forceinline__ float wave_sum(float v) {
#pragma unroll
  for (int o = 16; o > 0; o >>= 1) v += __shfl_xor(v, o, 32);
  return v;
}
static __device__ __forceinline__ float half_max(float v) {
#pragma unroll
  for (int o = 8; o > 0; o >>= 1) v = fmaxf(v, __shfl_xor(v, o, 32));
  return v;
}
static __device__ __forceinline__ float half_sum(float v) {
#pragma unroll
  for (int o = 8; o > 0; o >>= 1) v += __shfl_xor(v, o, 32);
  return v;
}

// ---------------- Weight prep: transpose to bf16 [n][k], fold 1/sqrt(C) into Wq.
__global__ void prep_w_kernel(const float* __restrict__ Wq, const float* __restrict__ Wk,
                              const float* __restrict__ Wv, const float* __restrict__ Wg,
                              const float* __restrict__ Wo, bf16* __restrict__ WT) {
  int idx = blockIdx.x * blockDim.x + threadIdx.x;   // 0 .. 5*65536-1
  int mat = idx >> 16;
  int e   = idx & 65535;
  int n   = e >> 8;
  int kk  = e & 255;
  const float* W = mat == 0 ? Wq : mat == 1 ? Wk : mat == 2 ? Wv : mat == 3 ? Wg : Wo;
  float sc = (mat == 0) ? 0.17677669529663687f : 1.0f;  // 1/sqrt(32)
  WT[idx] = (bf16)(W[kk * 256 + n] * sc);
}

// ---------------- msa LayerNorm -> bf16. One wave per token (8 channels/lane).
__global__ void ln_msa_kernel(const float* __restrict__ x, const float* __restrict__ g,
                              const float* __restrict__ b, bf16* __restrict__ y) {
  int wave = threadIdx.x >> 5, lane = threadIdx.x & 31;
  int t = blockIdx.x * 8 + wave;
  const float* xr = x + (size_t)t * 256 + lane * 8;
  float v[8];
#pragma unroll
  for (int e = 0; e < 8; ++e) v[e] = xr[e];
  float s = 0.0f;
#pragma unroll
  for (int e = 0; e < 8; ++e) s += v[e];
  float mu = wave_sum(s) * (1.0f / 256.0f);
  float d2 = 0.0f;
#pragma unroll
  for (int e = 0; e < 8; ++e) { float d = v[e] - mu; d2 += d * d; }
  float inv = rsqrtf(wave_sum(d2) * (1.0f / 256.0f) + 1e-5f);
  bf16* yr = y + (size_t)t * 256 + lane * 8;
#pragma unroll
  for (int e = 0; e < 8; ++e) {
    int c = lane * 8 + e;
    yr[e] = (bf16)((v[e] - mu) * inv * g[c] + b[c]);
  }
}

// ---------------- pair LayerNorm + W2d -> nb_bias TRANSPOSED: nbT[h][j][i] (f32).
__global__ void pair_nb_kernel(const float* __restrict__ pair, const float* __restrict__ g,
                               const float* __restrict__ b, const float* __restrict__ W2d,
                               float* __restrict__ nbT) {
  int wave = threadIdx.x >> 5, lane = threadIdx.x & 31;
  int p = blockIdx.x * 8 + wave;            // p = i*256 + j
  int i = p >> 8, j = p & 255;
  const float* xr = pair + (size_t)p * 128 + lane * 4;
  float v[4];
#pragma unroll
  for (int e = 0; e < 4; ++e) v[e] = xr[e];
  float s = v[0] + v[1] + v[2] + v[3];
  float mu = wave_sum(s) * (1.0f / 128.0f);
  float d2 = 0.0f;
#pragma unroll
  for (int e = 0; e < 4; ++e) { float d = v[e] - mu; d2 += d * d; }
  float inv = rsqrtf(wave_sum(d2) * (1.0f / 128.0f) + 1e-5f);
  float hacc[8] = {0, 0, 0, 0, 0, 0, 0, 0};
#pragma unroll
  for (int e = 0; e < 4; ++e) {
    int c = lane * 4 + e;
    float nv = (v[e] - mu) * inv * g[c] + b[c];
#pragma unroll
    for (int h = 0; h < 8; ++h) hacc[h] += nv * W2d[c * 8 + h];
  }
#pragma unroll
  for (int h = 0; h < 8; ++h) hacc[h] = wave_sum(hacc[h]);
  if (lane == 0) {
#pragma unroll
    for (int h = 0; h < 8; ++h)
      nbT[(size_t)h * 65536 + (size_t)j * 256 + i] = hacc[h];
  }
}

// ---------------- WMMA GEMM: [32768 x 256] (bf16) @ B^T[256][256] (bf16).
// B slice is staged once per block into LDS via the async copy engine.
// mode 0: bf16 [m][h][tok][32] (q/k)   mode 2: bf16 [m][h][32][tok] (v^T)
// mode 3: bf16 sigmoid(x+bias) row-major (gate)   mode 4: f32 x+bias (final out)
#define B_LD 264
__global__ void gemm_kernel(const bf16* __restrict__ A, const bf16* __restrict__ BT,
                            int mode, const float* __restrict__ bias, void* __restrict__ out) {
  __shared__ __align__(16) bf16 Bs[64 * B_LD];
  int wave = threadIdx.x >> 5, lane = threadIdx.x & 31;
  int row0 = (blockIdx.x * 8 + wave) * 16;
  int n0 = blockIdx.y * 64;
  // ---- async stage: 64 rows x 512B dense -> 528B-stride padded LDS
  unsigned bs0 = lds_off(Bs);
  for (int c = threadIdx.x; c < 2048; c += 256) {
    int rr = c >> 5, part = c & 31;
    async_ld_b128(bs0 + rr * (B_LD * 2) + part * 16,
                  BT + (size_t)(n0 + rr) * 256 + part * 8);
  }
  wait_async0();
  __syncthreads();

  v8f acc[4] = {{}, {}, {}, {}};
#pragma unroll 2
  for (int kk = 0; kk < 256; kk += 32) {
    v16bf af = load_frag(A + (size_t)row0 * 256, 256, kk);
#pragma unroll
    for (int t = 0; t < 4; ++t) {
      v16bf bfr = load_frag(Bs + t * 16 * B_LD, B_LD, kk);
      acc[t] = wmma_bf16(af, bfr, acc[t]);
    }
  }
  int colL = lane & 15, rbase = (lane >> 4) * 8;
#pragma unroll
  for (int t = 0; t < 4; ++t) {
    int n = n0 + t * 16 + colL;
    int h = n >> 5, c = n & 31;
#pragma unroll
    for (int r = 0; r < 8; ++r) {
      int row = row0 + rbase + r;
      int m = row >> 8, i = row & 255;
      float val = acc[t][r];
      if (mode == 0) {
        ((bf16*)out)[(((size_t)(m * 8 + h)) * 256 + i) * 32 + c] = (bf16)val;
      } else if (mode == 2) {
        ((bf16*)out)[(((size_t)(m * 8 + h)) * 32 + c) * 256 + i] = (bf16)val;
      } else if (mode == 3) {
        float z = val + bias[n];
        ((bf16*)out)[(size_t)row * 256 + n] = (bf16)(1.0f / (1.0f + __expf(-z)));
      } else {
        ((float*)out)[(size_t)row * 256 + n] = val + bias[n];
      }
    }
  }
}

// ---------------- Attention per (m,h): S = qk^T + bias, softmax, P@V, gate.
#define P_LD 264   // 528B stride -> conflict-free DS b128
#define K_LD 56    // 112B stride -> conflict-free DS b128
#define V_LD 264
__global__ void attn_kernel(const bf16* __restrict__ q, const bf16* __restrict__ k,
                            const bf16* __restrict__ vt, const float* __restrict__ nbT,
                            const float* __restrict__ mask, const bf16* __restrict__ gate,
                            bf16* __restrict__ outa) {
  __shared__ __align__(16) bf16 Pbuf[8][16 * P_LD];
  __shared__ __align__(16) bf16 Kb[256 * K_LD];
  __shared__ __align__(16) bf16 Vb[32 * V_LD];
  int m = blockIdx.x, h = blockIdx.y;
  int wave = threadIdx.x >> 5, lane = threadIdx.x & 31;
  int colL = lane & 15, rbase = (lane >> 4) * 8;
  const bf16* qh = q + (size_t)(m * 8 + h) * (256 * 32);
  const bf16* kh = k + (size_t)(m * 8 + h) * (256 * 32);
  const bf16* vh = vt + (size_t)(m * 8 + h) * (32 * 256);
  const float* nbh = nbT + (size_t)h * 65536;
  bf16* Pw = Pbuf[wave];

  // ---- async stage K (256x32 dense -> 112B stride) and V^T (32x256 -> 528B stride)
  unsigned kb0 = lds_off(Kb), vb0 = lds_off(Vb);
  for (int c = threadIdx.x; c < 1024; c += 256) {     // K: 4 chunks/row
    int rr = c >> 2, part = c & 3;
    async_ld_b128(kb0 + rr * (K_LD * 2) + part * 16, kh + rr * 32 + part * 8);
  }
  for (int c = threadIdx.x; c < 1024; c += 256) {     // V: 32 chunks/row
    int rr = c >> 5, part = c & 31;
    async_ld_b128(vb0 + rr * (V_LD * 2) + part * 16, vh + rr * 256 + part * 8);
  }
  wait_async0();
  __syncthreads();

  for (int iter = 0; iter < 2; ++iter) {
    int i0 = (iter * 8 + wave) * 16;
    // ---- S = q_tile @ k^T  (K=32 -> one WMMA per 16x16 j-tile)
    v16bf aq = load_frag(qh + (size_t)i0 * 32, 32, 0);
    v8f s[16];
#pragma unroll
    for (int jt = 0; jt < 16; ++jt) {
      v16bf bk = load_frag(Kb + jt * 16 * K_LD, K_LD, 0);
      v8f z = {};
      s[jt] = wmma_bf16(aq, bk, z);
    }
    // ---- pair bias (transposed: 8 contiguous floats per lane) + mask bias
#pragma unroll
    for (int jt = 0; jt < 16; ++jt) {
      int j = jt * 16 + colL;
      float mb = 1e9f * (mask[m * 256 + j] - 1.0f);
      const float* nbp = nbh + (size_t)j * 256 + i0 + rbase;
      float4 nA = ((const float4*)nbp)[0];
      float4 nB = ((const float4*)nbp)[1];
      s[jt][0] += nA.x + mb; s[jt][1] += nA.y + mb;
      s[jt][2] += nA.z + mb; s[jt][3] += nA.w + mb;
      s[jt][4] += nB.x + mb; s[jt][5] += nB.y + mb;
      s[jt][6] += nB.z + mb; s[jt][7] += nB.w + mb;
    }
    // ---- softmax along j
#pragma unroll
    for (int r = 0; r < 8; ++r) {
      float mx = -3.0e38f;
#pragma unroll
      for (int jt = 0; jt < 16; ++jt) mx = fmaxf(mx, s[jt][r]);
      mx = half_max(mx);
      float sum = 0.0f;
#pragma unroll
      for (int jt = 0; jt < 16; ++jt) {
        float e = __expf(s[jt][r] - mx);
        s[jt][r] = e;
        sum += e;
      }
      sum = half_sum(sum);
      float rs = 1.0f / sum;
#pragma unroll
      for (int jt = 0; jt < 16; ++jt)
        Pw[(rbase + r) * P_LD + jt * 16 + colL] = (bf16)(s[jt][r] * rs);
    }
    // ---- O = P @ V (A-frags from LDS; B-frags from staged V^T rows)
    v8f o0 = {}, o1 = {};
#pragma unroll
    for (int kc = 0; kc < 8; ++kc) {
      v16bf ap = load_frag(Pw, P_LD, kc * 32);
      v16bf b0 = load_frag(Vb, V_LD, kc * 32);
      v16bf b1 = load_frag(Vb + 16 * V_LD, V_LD, kc * 32);
      o0 = wmma_bf16(ap, b0, o0);
      o1 = wmma_bf16(ap, b1, o1);
    }
    // ---- gate and store bf16 (feeds final WMMA GEMM)
#pragma unroll
    for (int ct = 0; ct < 2; ++ct) {
      v8f oc = ct ? o1 : o0;
      int c = ct * 16 + colL;
#pragma unroll
      for (int r = 0; r < 8; ++r) {
        int i = i0 + rbase + r;
        size_t off = ((size_t)(m * 256 + i)) * 256 + h * 32 + c;
        outa[off] = (bf16)(oc[r] * (float)gate[off]);
      }
    }
  }
}

extern "C" void kernel_launch(void* const* d_in, const int* in_sizes, int n_in,
                              void* d_out, int out_size, void* d_ws, size_t ws_size,
                              hipStream_t stream) {
  const float* msa_act  = (const float*)d_in[0];
  const float* pair_act = (const float*)d_in[1];
  const float* msa_mask = (const float*)d_in[2];
  const float* qn_g = (const float*)d_in[3];
  const float* qn_b = (const float*)d_in[4];
  const float* pn_g = (const float*)d_in[5];
  const float* pn_b = (const float*)d_in[6];
  const float* Wq  = (const float*)d_in[7];
  const float* Wk  = (const float*)d_in[8];
  const float* Wv  = (const float*)d_in[9];
  const float* W2d = (const float*)d_in[10];
  const float* Wo  = (const float*)d_in[11];
  const float* bo  = (const float*)d_in[12];
  const float* Wg  = (const float*)d_in[13];
  const float* bg  = (const float*)d_in[14];
  (void)in_sizes; (void)n_in; (void)out_size; (void)ws_size;

  char* ws = (char*)d_ws;
  size_t off = 0;
  auto alloc = [&](size_t bytes) -> void* {
    void* p = ws + off;
    off += (bytes + 255) & ~(size_t)255;
    return p;
  };
  const size_t TOK = 32768;  // 128 * 256 tokens
  bf16*  msaln = (bf16*)alloc(TOK * 256 * 2);
  bf16*  WT    = (bf16*)alloc(5 * 65536 * 2);       // WqT,WkT,WvT,WgT,WoT
  float* nbT   = (float*)alloc(8 * 65536 * 4);      // [h][j][i]
  bf16*  qb    = (bf16*)alloc(TOK * 256 * 2);
  bf16*  kb    = (bf16*)alloc(TOK * 256 * 2);
  bf16*  vtb   = (bf16*)alloc(TOK * 256 * 2);
  bf16*  gateb = (bf16*)alloc(TOK * 256 * 2);
  bf16*  attnb = (bf16*)alloc(TOK * 256 * 2);

  prep_w_kernel<<<5 * 65536 / 256, 256, 0, stream>>>(Wq, Wk, Wv, Wg, Wo, WT);
  ln_msa_kernel<<<TOK / 8, 256, 0, stream>>>(msa_act, qn_g, qn_b, msaln);
  pair_nb_kernel<<<65536 / 8, 256, 0, stream>>>(pair_act, pn_g, pn_b, W2d, nbT);

  dim3 gg(256, 4);  // 128 rows/block x 8 waves; 4 col-groups of 64
  gemm_kernel<<<gg, 256, 0, stream>>>(msaln, WT + 0 * 65536, 0, nullptr, qb);
  gemm_kernel<<<gg, 256, 0, stream>>>(msaln, WT + 1 * 65536, 0, nullptr, kb);
  gemm_kernel<<<gg, 256, 0, stream>>>(msaln, WT + 2 * 65536, 2, nullptr, vtb);
  gemm_kernel<<<gg, 256, 0, stream>>>(msaln, WT + 3 * 65536, 3, bg, gateb);

  attn_kernel<<<dim3(128, 8), 256, 0, stream>>>(qb, kb, vtb, nbT, msa_mask, gateb, attnb);

  gemm_kernel<<<gg, 256, 0, stream>>>(attnb, WT + 4 * 65536, 4, bo, d_out);
}
